// Model_2190433321001
// MI455X (gfx1250) — matrix operand order repaired and checked by reading, not compiled, not run on hardware
//
#include <hip/hip_runtime.h>

typedef __attribute__((ext_vector_type(16))) _Float16 v16h;
typedef __attribute__((ext_vector_type(8)))  float    v8f;

#define L_LEN 16384
#define D_DIM 1000
#define CHUNK 128
#define NCHUNK (L_LEN / CHUNK)   /* 128 */

#if __has_builtin(__builtin_amdgcn_global_load_async_to_lds_b32)
#define HAVE_ASYNC_LDS 1
#else
#define HAVE_ASYNC_LDS 0
#endif

// ---------------------------------------------------------------------------
// Kernel 0: zero-init hRe accumulator (ws is poisoned 0xAA).
// ---------------------------------------------------------------------------
__global__ void k_zero(float* __restrict__ p, int n) {
  int i = blockIdx.x * blockDim.x + threadIdx.x;
  if (i < n) p[i] = 0.f;
}

// ---------------------------------------------------------------------------
// Kernel 1: blocked parallel scan.
// h_d(last) = sum_t x_t * z_d^{L-1-t},   z_d = exp(-exp(size_d)) * cis(theta_d)
// chunk (c,d): v = sum_{t in chunk} z^{chunkend-1-t} x_t  (forward recurrence)
// then rotate by z^{CHUNK*(NCHUNK-1-c)} in closed form, accumulate Re part.
// grid = (4, 128), block = 256. All threads of a block share one x-chunk,
// staged into LDS with the CDNA5 async-to-LDS path (ASYNCcnt).
// ---------------------------------------------------------------------------
__global__ __launch_bounds__(256) void k_scan(const float* __restrict__ x,
                                              const float* __restrict__ sz,
                                              const float* __restrict__ theta,
                                              float* __restrict__ hRe) {
  __shared__ float xs[CHUNK];
  const int tid = threadIdx.x;
  const int c   = blockIdx.y;
  const int t0  = c * CHUNK;

#if HAVE_ASYNC_LDS
  if (tid < CHUNK) {
    __builtin_amdgcn_global_load_async_to_lds_b32(
        (__attribute__((address_space(1))) int*)(x + t0 + tid),
        (__attribute__((address_space(3))) int*)&xs[tid], 0, 0);
  }
#if __has_builtin(__builtin_amdgcn_s_wait_asynccnt)
  __builtin_amdgcn_s_wait_asynccnt(0);
#else
  asm volatile("s_wait_asynccnt 0" ::: "memory");
#endif
#else
  if (tid < CHUNK) xs[tid] = x[t0 + tid];
#endif
  __syncthreads();

  const int d = blockIdx.x * blockDim.x + tid;
  if (d >= D_DIM) return;

  const float e  = __expf(sz[d]);      // exp(size) > 0
  const float th = theta[d];
  const float r  = __expf(-e);         // |z| in (0,1)
  const float zr = r * __cosf(th);
  const float zi = r * __sinf(th);

  float vr = 0.f, vi = 0.f;
  const float4* xs4 = (const float4*)xs;
#pragma unroll 4
  for (int i = 0; i < CHUNK / 4; ++i) {
    float4 xv = xs4[i];
#define STEP(xx)                                              \
    do {                                                      \
      float nvr = __fmaf_rn(zr, vr, __fmaf_rn(-zi, vi, xx));  \
      float nvi = __fmaf_rn(zr, vi, zi * vr);                 \
      vr = nvr; vi = nvi;                                     \
    } while (0)
    STEP(xv.x); STEP(xv.y); STEP(xv.z); STEP(xv.w);
#undef STEP
  }

  // w = z^n, n = CHUNK*(NCHUNK-1-c), closed form
  const float n   = (float)(CHUNK * (NCHUNK - 1 - c));
  const float mag = __expf(-n * e);    // underflows to 0 for fast-decay channels
  const float ang = n * th;
  const float wr  = mag * __cosf(ang);
  const float wi  = mag * __sinf(ang);

  atomicAdd(&hRe[d], wr * vr - wi * vi);
}

// ---------------------------------------------------------------------------
// Kernel 2: u = hRe @ W1^T + b1 ; x1 = relu(u + x[L-1])  via WMMA f16->f32.
// One wave (32 threads) per 16-column output tile; 63 blocks cover D=1000.
// A (16x32 f16): row 0 = hRe slice, rows 1..15 = 0. The row masking is done
// once via a zero page in LDS (lanes with m!=0 read zeros), so the k-loop is
// pure ds_load + v_wmma. B (32x16 f16) = W1^T tile staged in LDS (f32->f16).
// ---------------------------------------------------------------------------
__global__ __launch_bounds__(32) void k_gemv(const float* __restrict__ hRe,
                                             const float* __restrict__ W1,
                                             const float* __restrict__ b1,
                                             const float* __restrict__ x,
                                             float* __restrict__ x1) {
  __shared__ _Float16 hlds[2048];      // [0,1000) data, [1000,2048) zeros
  __shared__ _Float16 wlds[16 * 1024];
  const int lane = threadIdx.x;
  const int j0   = blockIdx.x * 16;

  for (int i = lane; i < 2048; i += 32)
    hlds[i] = (i < D_DIM) ? (_Float16)hRe[i] : (_Float16)0.f;

  for (int i = lane; i < 16 * 1024; i += 32) {
    const int row = i >> 10;
    const int k   = i & 1023;
    const int j   = j0 + row;
    const float w = (k < D_DIM && j < D_DIM) ? W1[j * D_DIM + k] : 0.f;
    wlds[i] = (_Float16)w;
  }
  __syncthreads();

  const int m  = lane & 15;   // output column n; only A-row m==0 carries data
  const int hi = lane >> 4;   // lane-group selector per WMMA VGPR layout
  // A-side base: row-0 lanes read real data, others read the zero page.
  const int apage = (m == 0) ? 0 : 1024;
  v8f acc = {};

  for (int kk = 0; kk < 32; ++kk) {
    const int k0 = kk * 32;
    v16h a, b;
    // A fragment (16-bit A 16x32): lanes 0-15 hold K {0..7,16..23},
    // lanes 16-31 hold K {8..15,24..31}.
    const int abase = apage + k0 + (hi ? 8 : 0);
#pragma unroll
    for (int t = 0; t < 8; ++t) {
      a[t]     = hlds[abase + t];
      a[8 + t] = hlds[abase + 16 + t];
    }
    // B fragment (32x16): lane n=m holds column n; lanes 0-15: K {0..7,16..23},
    // lanes 16-31: K {8..15,24..31}.  B[k][n] = W1[j0+n][k].
    const int bbase = m * 1024 + k0 + (hi ? 8 : 0);
#pragma unroll
    for (int t = 0; t < 8; ++t) {
      b[t]     = wlds[bbase + t];
      b[8 + t] = wlds[bbase + 16 + t];
    }

    acc = __builtin_amdgcn_wmma_f32_16x16x32_f16(
        /*neg_a=*/false, a, /*neg_b=*/false, b,
        /*c_mod=*/(short)0, acc, /*reuse_a=*/false, /*reuse_b=*/false);
  }

  // D row M=0 lives in acc[0] of lanes 0..15 (N = lane).
  const int j = j0 + lane;
  if (lane < 16 && j < D_DIM) {
    float u = acc[0] + b1[j] + x[L_LEN - 1];
    x1[j] = u > 0.f ? u : 0.f;
  }
}

// ---------------------------------------------------------------------------
// Kernel 3: logits = x1 @ Wf^T + bf ; softmax.  10 waves, one per logit.
// ---------------------------------------------------------------------------
__global__ __launch_bounds__(320) void k_final(const float* __restrict__ x1,
                                               const float* __restrict__ Wf,
                                               const float* __restrict__ bf,
                                               float* __restrict__ out) {
  __shared__ float lg[10];
  const int w    = threadIdx.x / 32;
  const int lane = threadIdx.x % 32;

  float p = 0.f;
  for (int j = lane; j < D_DIM; j += 32) p += x1[j] * Wf[w * D_DIM + j];
#pragma unroll
  for (int off = 16; off > 0; off >>= 1) p += __shfl_down(p, off, 32);
  if (lane == 0) lg[w] = p + bf[w];
  __syncthreads();

  if (threadIdx.x == 0) {
    float mx = lg[0];
    for (int i = 1; i < 10; ++i) mx = fmaxf(mx, lg[i]);
    float e[10], s = 0.f;
    for (int i = 0; i < 10; ++i) { e[i] = __expf(lg[i] - mx); s += e[i]; }
    for (int i = 0; i < 10; ++i) out[i] = e[i] / s;
  }
}

// ---------------------------------------------------------------------------
extern "C" void kernel_launch(void* const* d_in, const int* in_sizes, int n_in,
                              void* d_out, int out_size, void* d_ws, size_t ws_size,
                              hipStream_t stream) {
  const float* x     = (const float*)d_in[0];
  const float* sz    = (const float*)d_in[1];
  const float* theta = (const float*)d_in[2];
  const float* W1    = (const float*)d_in[3];
  const float* b1    = (const float*)d_in[4];
  const float* Wf    = (const float*)d_in[5];
  const float* bf    = (const float*)d_in[6];
  float* out = (float*)d_out;

  float* hRe = (float*)d_ws;                    // 1024 floats
  float* x1  = (float*)((char*)d_ws + 4096);    // 1000 floats

  k_zero<<<4, 256, 0, stream>>>(hRe, 1024);

  dim3 g1(4, NCHUNK);                           // 1024 channels x 128 chunks
  k_scan<<<g1, 256, 0, stream>>>(x, sz, theta, hRe);

  k_gemv<<<63, 32, 0, stream>>>(hRe, W1, b1, x, x1);

  k_final<<<1, 320, 0, stream>>>(x1, Wf, bf, out);
}
